// GPT_1769526526602
// MI455X (gfx1250) — matrix operand order repaired and checked by reading, not compiled, not run on hardware
//
#include <hip/hip_runtime.h>
#include <math.h>

// ---------------- types ----------------
typedef __bf16 bf16;
typedef __attribute__((ext_vector_type(16))) __bf16 v16bf;
typedef __attribute__((ext_vector_type(8)))  __bf16 v8bf;
typedef __attribute__((ext_vector_type(8)))  float  v8f;

// ---------------- model dims ----------------
static constexpr int Bb = 2;
static constexpr int S  = 1024;
static constexpr int D  = 768;
static constexpr int Hh = 12;
static constexpr int DK = 64;
static constexpr int Ll = 12;
static constexpr int F  = 3072;
static constexpr int V  = 50257;
static constexpr int M  = Bb * S;        // 2048 rows

// ---------------- fragment helpers (CDNA5 WMMA VGPR layouts) ----------------
__device__ inline v8f zero8() {
    v8f z;
#pragma unroll
    for (int i = 0; i < 8; ++i) z[i] = 0.0f;
    return z;
}

// A-fragment: elements 0..7 from lo[0..7], 8..15 from hi[0..7]
__device__ inline v16bf frag_split(const bf16* lo, const bf16* hi) {
    v8bf a = *(const v8bf*)lo;
    v8bf b = *(const v8bf*)hi;
    v16bf r;
#pragma unroll
    for (int i = 0; i < 8; ++i) { r[i] = a[i]; r[i + 8] = b[i]; }
    return r;
}

// B-fragment: 16 consecutive K values
__device__ inline v16bf frag16(const bf16* p) { return frag_split(p, p + 8); }

__device__ inline v8f wmma_bf16(v16bf a, v16bf b, v8f c) {
    return __builtin_amdgcn_wmma_f32_16x16x32_bf16(false, a, false, b, (short)0, c,
                                                   false, false);
}

// ---------------- embedding + sinusoidal PE ----------------
__global__ void embed_kernel(const int* __restrict__ x,
                             const float* __restrict__ tok,
                             float* __restrict__ h) {
    int row = blockIdx.x;              // 0..M-1
    int s = row & (S - 1);
    int tokid = x[row];
    const float scale = 27.712812921102035f;          // sqrt(768)
    const float c = 9.210340371976184f / (float)D;    // ln(10000)/D
    for (int i = threadIdx.x; i < D; i += blockDim.x) {
        int p = i >> 1;
        float div = __expf(-(float)(2 * p) * c);
        float ang = (float)s * div;
        float pe = (i & 1) ? __cosf(ang) : __sinf(ang);
        h[(long)row * D + i] = tok[(long)tokid * D + i] * scale + pe;
    }
}

// ---------------- layernorm (fp32 in -> bf16 out) ----------------
__global__ void ln_kernel(const float* __restrict__ h,
                          const float* __restrict__ g,
                          const float* __restrict__ b,
                          bf16* __restrict__ y) {
    __shared__ float red[256];
    int row = blockIdx.x;
    int t = threadIdx.x;
    const float* hr = h + (long)row * D;
    float s = 0.f, s2 = 0.f;
    for (int i = t; i < D; i += 256) { float v = hr[i]; s += v; s2 += v * v; }
    red[t] = s; __syncthreads();
    for (int o = 128; o > 0; o >>= 1) { if (t < o) red[t] += red[t + o]; __syncthreads(); }
    float mean = red[0] / (float)D;
    __syncthreads();
    red[t] = s2; __syncthreads();
    for (int o = 128; o > 0; o >>= 1) { if (t < o) red[t] += red[t + o]; __syncthreads(); }
    float var = red[0] / (float)D - mean * mean;
    float inv = rsqrtf(var + 1e-5f);
    for (int i = t; i < D; i += 256) {
        float v = (hr[i] - mean) * inv * g[i] + b[i];
        y[(long)row * D + i] = (bf16)v;
    }
}

// ---------------- weight transpose + fp32->bf16 convert ----------------
// src is K x N (row-major fp32); dst is N x K (row-major bf16)
__global__ void transpose_bf16_kernel(const float* __restrict__ src,
                                      bf16* __restrict__ dst, int K, int N) {
    long idx = (long)blockIdx.x * blockDim.x + threadIdx.x;
    long total = (long)K * N;
    if (idx >= total) return;
    long n = idx / K, k = idx % K;
    dst[idx] = (bf16)src[k * (long)N + n];
}

// ---------------- WMMA GEMM ----------------
// C[M,N] = A[M,K](bf16) x BT[N,K](bf16)^T + bias, then per-mode epilogue:
//   mode 0: fp32 out, optional residual add
//   mode 1: bf16 out row-major (with optional relu)
//   mode 2: bf16 out reshaped to (B,H,S,DK)   [Q/K layout]
//   mode 3: bf16 out reshaped to (B,H,DK,S)   [V^T layout]
// Out-of-range N columns: B row index is CLAMPED (no EXEC divergence in the
// K loop); the garbage accumulators for those columns are discarded by the
// guarded store epilogue.
__global__ __launch_bounds__(128) void gemm_wmma_kernel(
    const bf16* __restrict__ A, const bf16* __restrict__ BT,
    const float* __restrict__ bias, const float* __restrict__ residual,
    float* __restrict__ outF, bf16* __restrict__ outB,
    int Mdim, int Ndim, int Kdim, int mode, int relu) {
    int lane = threadIdx.x & 31;
    int w = threadIdx.x >> 5;
    int wm = w >> 1, wn = w & 1;
    int m0 = blockIdx.y * 128 + wm * 64;
    int n0 = blockIdx.x * 128 + wn * 64;
    int hi = lane >> 4;
    int l15 = lane & 15;
    int ka = hi ? 8 : 0;
    int kb = hi ? 8 : 0;

    // hoisted per-fragment base pointers
    const bf16* ap[4];
    const bf16* bp[4];
#pragma unroll
    for (int i = 0; i < 4; ++i)
        ap[i] = A + (long)(m0 + 16 * i + l15) * Kdim + ka;
#pragma unroll
    for (int j = 0; j < 4; ++j) {
        int n = n0 + 16 * j + l15;
        int nc = (n < Ndim) ? n : (Ndim - 1);   // clamp: keep loads unconditional
        bp[j] = BT + (long)nc * Kdim + kb;
    }

    v8f acc[4][4];
#pragma unroll
    for (int i = 0; i < 4; ++i)
#pragma unroll
        for (int j = 0; j < 4; ++j) acc[i][j] = zero8();

    for (int kk = 0; kk < Kdim; kk += 32) {
        // prefetch the streaming weight side ~512B ahead (global_prefetch_b8)
#pragma unroll
        for (int j = 0; j < 4; ++j) __builtin_prefetch(bp[j] + kk + 256, 0, 3);

        v16bf af[4], bfr[4];
#pragma unroll
        for (int i = 0; i < 4; ++i)
            af[i] = frag_split(ap[i] + kk, ap[i] + kk + 16);
#pragma unroll
        for (int j = 0; j < 4; ++j)
            bfr[j] = frag_split(bp[j] + kk, bp[j] + kk + 16);
#pragma unroll
        for (int i = 0; i < 4; ++i)
#pragma unroll
            for (int j = 0; j < 4; ++j) acc[i][j] = wmma_bf16(af[i], bfr[j], acc[i][j]);
    }

#pragma unroll
    for (int i = 0; i < 4; ++i) {
#pragma unroll
        for (int j = 0; j < 4; ++j) {
            int n = n0 + 16 * j + l15;
            if (n >= Ndim) continue;
            float bs = bias ? bias[n] : 0.0f;
#pragma unroll
            for (int r = 0; r < 8; ++r) {
                int m = m0 + 16 * i + 8 * hi + r;
                if (m >= Mdim) continue;
                float v = acc[i][j][r] + bs;
                if (relu) v = fmaxf(v, 0.0f);
                if (mode == 0) {
                    long idx = (long)m * Ndim + n;
                    if (residual) v += residual[idx];
                    outF[idx] = v;
                } else if (mode == 1) {
                    outB[(long)m * Ndim + n] = (bf16)v;
                } else if (mode == 2) {
                    int bi = m >> 10, s = m & (S - 1), hd = n >> 6, dk = n & 63;
                    outB[(((long)(bi * Hh + hd)) * S + s) * DK + dk] = (bf16)v;
                } else {
                    int bi = m >> 10, s = m & (S - 1), hd = n >> 6, dk = n & 63;
                    outB[(((long)(bi * Hh + hd)) * DK + dk) * S + s] = (bf16)v;
                }
            }
        }
    }
}

// ---------------- fused attention ----------------
// One wave per (batch, head, 16-row q tile).
// Q,K: (B,H,S,DK) bf16 ; VT: (B,H,DK,S) bf16 ; out: (B,S,D) bf16.
// Reproduces the reference's mask bug: positions k <= q are filled with -1e9.
__global__ __launch_bounds__(32) void attention_kernel(
    const bf16* __restrict__ Q, const bf16* __restrict__ Kt,
    const bf16* __restrict__ VT, bf16* __restrict__ attout) {
    __shared__ bf16 sc[16 * S];       // 32 KB: scores, then exp values
    __shared__ float smax[16];
    __shared__ float ssum[16];

    int lane = threadIdx.x;
    int wg = blockIdx.x;              // ((b*H + h)*64 + qt)
    int qt = wg & 63;
    int hd = (wg >> 6) % Hh;
    int bi = wg / (64 * Hh);
    const bf16* Qh = Q + ((long)(bi * Hh + hd)) * S * DK;
    const bf16* Kh = Kt + ((long)(bi * Hh + hd)) * S * DK;
    const bf16* Vh = VT + ((long)(bi * Hh + hd)) * DK * S;
    int hi = lane >> 4;
    int l15 = lane & 15;
    int rbase = 8 * hi;

    // Q tile fragments (K-dim = 64 -> two 32-chunks)
    v16bf aq[2];
    {
        int m = qt * 16 + l15;
#pragma unroll
        for (int t = 0; t < 2; ++t) {
            const bf16* p = Qh + (long)m * DK + 32 * t + (hi ? 8 : 0);
            aq[t] = frag_split(p, p + 16);
        }
    }

    // ---- pass 1: scores = QK^T, apply (buggy) mask, track row maxima ----
    float rmax[8];
#pragma unroll
    for (int r = 0; r < 8; ++r) rmax[r] = -3.0e38f;

    for (int ct = 0; ct < S / 16; ++ct) {
        v8f sacc = zero8();
        int n = ct * 16 + l15;
#pragma unroll
        for (int t = 0; t < 2; ++t) {
            const bf16* kp = Kh + (long)n * DK + 32 * t + (hi ? 8 : 0);
            v16bf bk = frag_split(kp, kp + 16);
            sacc = wmma_bf16(aq[t], bk, sacc);
        }
        int col = ct * 16 + l15;
#pragma unroll
        for (int r = 0; r < 8; ++r) {
            int qrow = qt * 16 + rbase + r;
            float vv = (col <= qrow) ? -1.0e9f : sacc[r] * 0.125f; // 1/sqrt(64)
            sc[(rbase + r) * S + col] = (bf16)vv;
            rmax[r] = fmaxf(rmax[r], vv);
        }
    }
    // reduce maxima across the 16 lanes of each half-wave
#pragma unroll
    for (int d = 1; d < 16; d <<= 1)
#pragma unroll
        for (int r = 0; r < 8; ++r)
            rmax[r] = fmaxf(rmax[r], __shfl_xor(rmax[r], d, 32));
    if (l15 == 0) {
#pragma unroll
        for (int r = 0; r < 8; ++r) smax[rbase + r] = rmax[r];
    }
    __syncthreads();

    // ---- pass 2: exp + row sums (2 lanes per row) ----
    {
        int row = lane >> 1;
        int half = lane & 1;
        float mx = smax[row];
        float sum = 0.0f;
        bf16* rp = sc + row * S + half * (S / 2);
        for (int j = 0; j < S / 2; ++j) {
            float e = __expf((float)rp[j] - mx);
            sum += e;
            rp[j] = (bf16)e;
        }
        sum += __shfl_xor(sum, 1, 32);
        if (half == 0) ssum[row] = sum;
    }
    __syncthreads();

    // ---- pass 3: O = P @ V (probabilities from LDS, V^T from global) ----
    v8f o[4];
#pragma unroll
    for (int j = 0; j < 4; ++j) o[j] = zero8();

    for (int kk = 0; kk < S; kk += 32) {
        const bf16* p = sc + l15 * S + kk + (hi ? 8 : 0);
        v16bf ap = frag_split(p, p + 16);
#pragma unroll
        for (int j = 0; j < 4; ++j) {
            const bf16* vp = Vh + (long)(16 * j + l15) * S + kk + (hi ? 8 : 0);
            v16bf bv = frag_split(vp, vp + 16);
            o[j] = wmma_bf16(ap, bv, o[j]);
        }
    }

#pragma unroll
    for (int j = 0; j < 4; ++j) {
#pragma unroll
        for (int r = 0; r < 8; ++r) {
            int rowl = rbase + r;
            int q = qt * 16 + rowl;
            int dk = 16 * j + l15;
            float inv = 1.0f / ssum[rowl];
            attout[((long)(bi * S + q)) * D + hd * DK + dk] = (bf16)(o[j][r] * inv);
        }
    }
}

// ---------------- host orchestration ----------------
extern "C" void kernel_launch(void* const* d_in, const int* in_sizes, int n_in,
                              void* d_out, int out_size, void* d_ws, size_t ws_size,
                              hipStream_t stream) {
    (void)in_sizes; (void)n_in; (void)out_size; (void)ws_size;
    const int*   x      = (const int*)d_in[0];
    const float* tok    = (const float*)d_in[1];
    const float* Wq     = (const float*)d_in[2];
    const float* bq     = (const float*)d_in[3];
    const float* Wk     = (const float*)d_in[4];
    const float* bk     = (const float*)d_in[5];
    const float* Wv     = (const float*)d_in[6];
    const float* bv     = (const float*)d_in[7];
    const float* Wo     = (const float*)d_in[8];
    const float* bo     = (const float*)d_in[9];
    const float* ln1g   = (const float*)d_in[10];
    const float* ln1b   = (const float*)d_in[11];
    const float* ln2g   = (const float*)d_in[12];
    const float* ln2b   = (const float*)d_in[13];
    const float* W1     = (const float*)d_in[14];
    const float* b1     = (const float*)d_in[15];
    const float* W2     = (const float*)d_in[16];
    const float* b2     = (const float*)d_in[17];
    const float* lnfg   = (const float*)d_in[18];
    const float* lnfb   = (const float*)d_in[19];
    const float* Wout   = (const float*)d_in[20];
    const float* bout   = (const float*)d_in[21];
    float* out = (float*)d_out;

    // ---- workspace layout (256B aligned) ----
    char* ws = (char*)d_ws;
    size_t cur = 0;
    auto alloc = [&](size_t bytes) -> char* {
        char* p = ws + cur;
        cur += (bytes + 255) & ~(size_t)255;
        return p;
    };
    const size_t szDD = (size_t)Ll * D * D;
    const size_t szDF = (size_t)Ll * D * F;
    bf16* WqT   = (bf16*)alloc(szDD * 2);
    bf16* WkT   = (bf16*)alloc(szDD * 2);
    bf16* WvT   = (bf16*)alloc(szDD * 2);
    bf16* WoT   = (bf16*)alloc(szDD * 2);
    bf16* W1T   = (bf16*)alloc(szDF * 2);
    bf16* W2T   = (bf16*)alloc(szDF * 2);
    bf16* WoutT = (bf16*)alloc((size_t)V * D * 2);
    float* h    = (float*)alloc((size_t)M * D * 4);
    bf16* ybf   = (bf16*)alloc((size_t)M * D * 2);
    bf16* qbf   = (bf16*)alloc((size_t)M * D * 2);
    bf16* kbf   = (bf16*)alloc((size_t)M * D * 2);
    bf16* vtbf  = (bf16*)alloc((size_t)M * D * 2);
    bf16* attbf = (bf16*)alloc((size_t)M * D * 2);
    bf16* actbf = (bf16*)alloc((size_t)M * F * 2);

    // ---- weight convert + transpose (bf16, N x K) ----
    auto launch_tr = [&](const float* src, bf16* dst, int K, int N) {
        long total = (long)K * N;
        int grid = (int)((total + 255) / 256);
        transpose_bf16_kernel<<<grid, 256, 0, stream>>>(src, dst, K, N);
    };
    for (int l = 0; l < Ll; ++l) {
        launch_tr(Wq + (size_t)l * D * D, WqT + (size_t)l * D * D, D, D);
        launch_tr(Wk + (size_t)l * D * D, WkT + (size_t)l * D * D, D, D);
        launch_tr(Wv + (size_t)l * D * D, WvT + (size_t)l * D * D, D, D);
        launch_tr(Wo + (size_t)l * D * D, WoT + (size_t)l * D * D, D, D);
        launch_tr(W1 + (size_t)l * D * F, W1T + (size_t)l * D * F, D, F);
        launch_tr(W2 + (size_t)l * F * D, W2T + (size_t)l * F * D, F, D);
    }
    launch_tr(Wout, WoutT, D, V);

    auto launch_gemm = [&](const bf16* A, const bf16* BT, const float* bias,
                           const float* residual, float* outF, bf16* outB,
                           int Md, int Nd, int Kd, int mode, int relu) {
        dim3 grid((Nd + 127) / 128, (Md + 127) / 128);
        gemm_wmma_kernel<<<grid, 128, 0, stream>>>(A, BT, bias, residual, outF, outB,
                                                   Md, Nd, Kd, mode, relu);
    };

    // ---- embedding ----
    embed_kernel<<<M, 256, 0, stream>>>(x, tok, h);

    // ---- transformer blocks ----
    for (int l = 0; l < Ll; ++l) {
        ln_kernel<<<M, 256, 0, stream>>>(h, ln1g + (size_t)l * D, ln1b + (size_t)l * D, ybf);
        launch_gemm(ybf, WqT + (size_t)l * D * D, bq + (size_t)l * D, nullptr,
                    nullptr, qbf, M, D, D, /*mode=*/2, 0);
        launch_gemm(ybf, WkT + (size_t)l * D * D, bk + (size_t)l * D, nullptr,
                    nullptr, kbf, M, D, D, /*mode=*/2, 0);
        launch_gemm(ybf, WvT + (size_t)l * D * D, bv + (size_t)l * D, nullptr,
                    nullptr, vtbf, M, D, D, /*mode=*/3, 0);
        attention_kernel<<<Bb * Hh * (S / 16), 32, 0, stream>>>(qbf, kbf, vtbf, attbf);
        launch_gemm(attbf, WoT + (size_t)l * D * D, bo + (size_t)l * D, h,
                    h, nullptr, M, D, D, /*mode=*/0, 0);
        ln_kernel<<<M, 256, 0, stream>>>(h, ln2g + (size_t)l * D, ln2b + (size_t)l * D, ybf);
        launch_gemm(ybf, W1T + (size_t)l * D * F, b1 + (size_t)l * F, nullptr,
                    nullptr, actbf, M, F, D, /*mode=*/1, /*relu=*/1);
        launch_gemm(actbf, W2T + (size_t)l * F * D, b2 + (size_t)l * D, h,
                    h, nullptr, M, D, F, /*mode=*/0, 0);
    }

    // ---- final LN + logits ----
    ln_kernel<<<M, 256, 0, stream>>>(h, lnfg, lnfb, ybf);
    launch_gemm(ybf, WoutT, bout, nullptr, out, nullptr, M, V, D, /*mode=*/0, 0);
}
